// DynamicVisible2InvisibleAttention_33097017983148
// MI455X (gfx1250) — matrix operand order repaired and checked by reading, not compiled, not run on hardware
//
#include <hip/hip_runtime.h>

typedef __attribute__((ext_vector_type(16))) __bf16 v16bf;
typedef __attribute__((ext_vector_type(8)))  __bf16 v8bf;
typedef __attribute__((ext_vector_type(8)))  float  v8f;

#define HW 4096
#define NB 8          // batch

__device__ __forceinline__ v16bf join8(v8bf lo, v8bf hi) {
  v16bf r;
#pragma unroll
  for (int i = 0; i < 8; ++i) { r[i] = lo[i]; r[8 + i] = hi[i]; }
  return r;
}

// ---------------------------------------------------------------------------
// Zero guard page (boundary taps load from here instead of branching)
// ---------------------------------------------------------------------------
__global__ void zero_guard_kernel(unsigned* __restrict__ g) {
  g[threadIdx.x] = 0u;          // <<<1,64>>> -> 256 B of zeros
}

// ---------------------------------------------------------------------------
// Cast x (fp32, channel-major) -> xcat blocks 0..7 (bf16, channels-last x32)
// xcat layout: [b][16][hw][32]
// ---------------------------------------------------------------------------
__global__ void __launch_bounds__(256) cast_x_kernel(const float* __restrict__ x,
                                                     __bf16* __restrict__ xcat) {
  int tid = blockIdx.x * blockDim.x + threadIdx.x;
  if (tid >= NB * 8 * HW * 32) return;
  int b   = tid >> 20;                 // 8*4096*32 = 2^20 per batch
  int rem = tid & ((1 << 20) - 1);
  int cb  = rem >> 17;                 // 4096*32 = 2^17 per block
  int p   = (rem >> 5) & 4095;
  int cl  = rem & 31;
  int ci  = cb * 32 + cl;
  float v = x[((size_t)(b * 256 + ci)) * HW + p];
  xcat[(((size_t)b * 16 + cb) * HW + p) * 32 + cl] = (__bf16)v;
}

// ---------------------------------------------------------------------------
// Repack OIHW fp32 weights into WMMA A-fragment layout:
// wpack[tap][cb][mt][lane 32][16 halves]  (32 contiguous bytes per lane)
// A 16x32 bf16 order used here (contiguous k, B matches): lane m = lane&15, base = (lane<16)?0:8,
//   half h: k = 2*base + h
// ---------------------------------------------------------------------------
__global__ void __launch_bounds__(256) repack_w_kernel(const float* __restrict__ W,
                                                       __bf16* __restrict__ wpack,
                                                       int Cin, int Cout) {
  int CB = Cin >> 5, MT = Cout >> 4;
  int total = 9 * CB * MT * 32;
  int tid = blockIdx.x * blockDim.x + threadIdx.x;
  if (tid >= total) return;
  int lane = tid & 31;
  int t2 = tid >> 5;
  int mt = t2 % MT; t2 /= MT;
  int cb = t2 % CB;
  int tap = t2 / CB;
  int m = lane & 15;
  int base = (lane & 16) ? 8 : 0;
#pragma unroll
  for (int h = 0; h < 16; ++h) {
    int k  = 2 * base + h;
    int ci = cb * 32 + k;
    int co = mt * 16 + m;
    wpack[(size_t)tid * 16 + h] = (__bf16)W[((size_t)co * Cin + ci) * 9 + tap];
  }
}

// ---------------------------------------------------------------------------
// dm = clamp(conv3x3(mask, Wd), 0, 1)   (Cout=1, tiny -> plain VALU kernel)
// ---------------------------------------------------------------------------
__global__ void __launch_bounds__(256) conv_dm_kernel(const float* __restrict__ mask,
                                                      const float* __restrict__ Wd,
                                                      float* __restrict__ dmOut) {
  int tid = blockIdx.x * blockDim.x + threadIdx.x;
  if (tid >= NB * HW) return;
  int b = tid >> 12;
  int p = tid & 4095;
  int y = p >> 6, x = p & 63;
  float sum = 0.f;
  const float* mb = mask + (size_t)b * 256 * HW;
  for (int ci = 0; ci < 256; ++ci) {
    const float* mrow = mb + (size_t)ci * HW;
    const float* wrow = Wd + ci * 9;
#pragma unroll
    for (int ky = 0; ky < 3; ++ky) {
      int yy = y + ky - 1;
      if (yy < 0 || yy >= 64) continue;
#pragma unroll
      for (int kx = 0; kx < 3; ++kx) {
        int xx = x + kx - 1;
        if (xx < 0 || xx >= 64) continue;
        sum = fmaf(mrow[yy * 64 + xx], wrow[ky * 3 + kx], sum);
      }
    }
  }
  dmOut[tid] = fminf(fmaxf(sum, 0.f), 1.f);
}

// ---------------------------------------------------------------------------
// Implicit-GEMM 3x3 SAME conv via WMMA, 2x2 register blocking:
// one wave computes a 32(co) x 32(p) tile with 4 accumulators, so each
// (A0,A1,B0,B1) load quad feeds 4 WMMAs. Boundary taps read a zero guard
// page via per-lane address select (no EXEC divergence).
// mode 0: channels-last bf16 out[b][hw][32]            (q, k ; Cout=32)
// mode 1: channel-major bf16 out[b][Cout][hw]          (v)
// mode 2: channel-major fp32 out[b][Cout][hw]          (final output)
// ---------------------------------------------------------------------------
__global__ void __launch_bounds__(128)
conv3x3_wmma(const __bf16* __restrict__ xcat,   // [b][16][hw][32]
             const __bf16* __restrict__ wpack,  // [9][CB][MT][32][16]
             const float*  __restrict__ bias,   // [Cout] or nullptr
             const __bf16* __restrict__ zguard, // 256B of zeros
             void* __restrict__ outp,
             int CB, int MT, int mode) {
  const int lane = threadIdx.x & 31;
  const int wid  = threadIdx.x >> 5;
  const int gid  = blockIdx.x * 4 + wid;
  const int MT2  = MT >> 1;
  const int tilesPerB = MT2 * (HW / 32);
  if (gid >= NB * tilesPerB) return;       // whole-wave uniform exit
  const int b  = gid / tilesPerB;
  int rem = gid - b * tilesPerB;
  const int mt2 = rem >> 7;                // HW/32 == 128
  const int pt2 = rem & 127;
  const int p0 = pt2 * 32;
  const int y0 = p0 >> 6;
  const int x0 = p0 & 63;                  // 0 or 32
  const int hi = lane >> 4;
  const int ln = lane & 15;
  const int kb = hi ? 16 : 0;

  v8f acc[2][2];
#pragma unroll
  for (int mi = 0; mi < 2; ++mi) {
    v8f c = {};
    if (bias) {
#pragma unroll
      for (int r = 0; r < 8; ++r) c[r] = bias[mt2 * 32 + mi * 16 + r + 8 * hi];
    }
    acc[mi][0] = c;
    acc[mi][1] = c;
  }
  const __bf16* xb = xcat + (size_t)b * 16 * HW * 32;

  for (int tap = 0; tap < 9; ++tap) {
    const int dy = tap / 3 - 1, dx = tap % 3 - 1;
    const int yy = y0 + dy;
    const bool rowok = (yy >= 0) && (yy < 64);
    const int xx0 = x0 + ln + dx;          // [-1, 16] or [31, 48]
    const int xx1 = xx0 + 16;              // always >= 15
    const bool v0ok = rowok && (xx0 >= 0) && (xx0 < 64);
    const bool v1ok = rowok && (xx1 < 64);
    const int pp0 = yy * 64 + xx0;
    const int pp1 = pp0 + 16;
    const __bf16* wtap = wpack + (size_t)tap * CB * MT * 512;
    for (int cb = 0; cb < CB; ++cb) {
      const __bf16* ap = wtap + (((size_t)(cb * MT + 2 * mt2)) * 32 + lane) * 16;
      v16bf a0 = *(const v16bf*)ap;
      v16bf a1 = *(const v16bf*)(ap + 512);
      const __bf16* bbase = xb + (size_t)cb * HW * 32 + kb;
      const __bf16* bp0 = v0ok ? (bbase + (size_t)pp0 * 32) : zguard;
      const __bf16* bp1 = v1ok ? (bbase + (size_t)pp1 * 32) : zguard;
      v16bf b0 = *(const v16bf*)bp0;
      v16bf b1 = *(const v16bf*)bp1;
      acc[0][0] = __builtin_amdgcn_wmma_f32_16x16x32_bf16(false, a0, false, b0,
                                                          (short)0, acc[0][0], false, false);
      acc[0][1] = __builtin_amdgcn_wmma_f32_16x16x32_bf16(false, a0, false, b1,
                                                          (short)0, acc[0][1], false, false);
      acc[1][0] = __builtin_amdgcn_wmma_f32_16x16x32_bf16(false, a1, false, b0,
                                                          (short)0, acc[1][0], false, false);
      acc[1][1] = __builtin_amdgcn_wmma_f32_16x16x32_bf16(false, a1, false, b1,
                                                          (short)0, acc[1][1], false, false);
    }
  }

  if (mode == 0) {                       // channels-last bf16 (record of 32)
    __bf16* out = (__bf16*)outp;
#pragma unroll
    for (int mi = 0; mi < 2; ++mi) {
#pragma unroll
      for (int ni = 0; ni < 2; ++ni) {
        const int p = p0 + ni * 16 + ln;
#pragma unroll
        for (int r = 0; r < 8; r += 2) {
          int co = mt2 * 32 + mi * 16 + r + 8 * hi;   // even
          union { __bf16 h[2]; unsigned u; } pk;
          pk.h[0] = (__bf16)acc[mi][ni][r];
          pk.h[1] = (__bf16)acc[mi][ni][r + 1];
          *(unsigned*)(out + (((size_t)b * HW + p) * 32 + co)) = pk.u;
        }
      }
    }
  } else if (mode == 1) {                // channel-major bf16
    __bf16* out = (__bf16*)outp;
    const int Cout = MT * 16;
#pragma unroll
    for (int mi = 0; mi < 2; ++mi) {
#pragma unroll
      for (int ni = 0; ni < 2; ++ni) {
        const int p = p0 + ni * 16 + ln;
#pragma unroll
        for (int r = 0; r < 8; ++r) {
          int co = mt2 * 32 + mi * 16 + r + 8 * hi;
          out[((size_t)(b * Cout + co)) * HW + p] = (__bf16)acc[mi][ni][r];
        }
      }
    }
  } else {                               // channel-major fp32
    float* out = (float*)outp;
    const int Cout = MT * 16;
#pragma unroll
    for (int mi = 0; mi < 2; ++mi) {
#pragma unroll
      for (int ni = 0; ni < 2; ++ni) {
        const int p = p0 + ni * 16 + ln;
#pragma unroll
        for (int r = 0; r < 8; ++r) {
          int co = mt2 * 32 + mi * 16 + r + 8 * hi;
          out[((size_t)(b * Cout + co)) * HW + p] = acc[mi][ni][r];
        }
      }
    }
  }
}

// ---------------------------------------------------------------------------
// Fused masked attention (flash-attention style), one wave per 16-row j tile.
// energy[j,l] = (q@k)[j,l] * mm[l] * im[j];  P = softmax_l;  O[c,j] = V @ P^T.
// Output written as bf16 into xcat blocks 8..15 (channels-last x32) so the
// final conv reads the [x, out] concat from a single buffer.
// ---------------------------------------------------------------------------
__global__ void __launch_bounds__(128)
attn_wmma(const __bf16* __restrict__ qp,   // [b][hw][32]
          const __bf16* __restrict__ kp,   // [b][hw][32]
          const __bf16* __restrict__ vc,   // [b][256][hw]
          const float*  __restrict__ dm,   // [b][hw]
          __bf16* __restrict__ xcat) {     // [b][16][hw][32]
  __shared__ __align__(64) __bf16 Plds[4][2][16][16];   // per-wave P tiles
  __shared__ float Xlds[4][16];                         // row<->col exchange
  const int lane = threadIdx.x & 31;
  const int wid  = threadIdx.x >> 5;
  const int gid  = blockIdx.x * 4 + wid;                // NB*256 tiles total
  const int b  = gid >> 8;
  const int j0 = (gid & 255) * 16;
  const int hi = lane >> 4;
  const int ln = lane & 15;
  const int abase = hi ? 8 : 0;
  const int kb = hi ? 16 : 0;

  // q A-fragment (16 j x 32 cqk), loaded once
  const __bf16* qrow = qp + ((size_t)b * HW + (j0 + ln)) * 32;
  v16bf qa = *(const v16bf*)(qrow + kb);

  const float* dmb = dm + (size_t)b * HW;
  float im8[8];
#pragma unroll
  for (int r = 0; r < 8; ++r) im8[r] = 1.0f - dmb[j0 + r + 8 * hi];

  v8f O[16];
#pragma unroll
  for (int t = 0; t < 16; ++t) O[t] = (v8f){};
  float m8[8], l8[8];
#pragma unroll
  for (int r = 0; r < 8; ++r) { m8[r] = -1e30f; l8[r] = 0.0f; }

  const __bf16* vb  = vc + (size_t)b * 256 * HW;
  const __bf16* kpb = kp + (size_t)b * HW * 32;

  for (int l0 = 0; l0 < HW; l0 += 32) {
    if (l0 + 32 < HW)
      __builtin_prefetch(kpb + (size_t)(l0 + 32 + ln) * 32, 0, 0);

    v16bf kf0 = *(const v16bf*)(kpb + ((size_t)(l0 + ln)) * 32 + kb);
    v16bf kf1 = *(const v16bf*)(kpb + ((size_t)(l0 + 16 + ln)) * 32 + kb);
    float mm0 = dmb[l0 + ln];
    float mm1 = dmb[l0 + 16 + ln];

    v8f z = {};
    v8f S0 = __builtin_amdgcn_wmma_f32_16x16x32_bf16(false, qa, false, kf0,
                                                     (short)0, z, false, false);
    v8f S1 = __builtin_amdgcn_wmma_f32_16x16x32_bf16(false, qa, false, kf1,
                                                     (short)0, z, false, false);
    float e0[8], e1[8], mx[8];
#pragma unroll
    for (int r = 0; r < 8; ++r) {
      e0[r] = S0[r] * (mm0 * im8[r]);
      e1[r] = S1[r] * (mm1 * im8[r]);
      mx[r] = fmaxf(e0[r], e1[r]);
    }
#pragma unroll
    for (int d = 1; d < 16; d <<= 1) {
#pragma unroll
      for (int r = 0; r < 8; ++r) mx[r] = fmaxf(mx[r], __shfl_xor(mx[r], d, 16));
    }
    float al[8], rs[8];
#pragma unroll
    for (int r = 0; r < 8; ++r) {
      float mn = fmaxf(m8[r], mx[r]);
      al[r] = __expf(m8[r] - mn);
      e0[r] = __expf(e0[r] - mn);
      e1[r] = __expf(e1[r] - mn);
      m8[r] = mn;
      rs[r] = e0[r] + e1[r];
    }
#pragma unroll
    for (int d = 1; d < 16; d <<= 1) {
#pragma unroll
      for (int r = 0; r < 8; ++r) rs[r] += __shfl_xor(rs[r], d, 16);
    }
#pragma unroll
    for (int r = 0; r < 8; ++r) l8[r] = l8[r] * al[r] + rs[r];

    // stage P (bf16) + row alphas through per-wave LDS (in-order DS ops)
#pragma unroll
    for (int r = 0; r < 8; ++r) {
      Plds[wid][0][r + 8 * hi][ln] = (__bf16)e0[r];
      Plds[wid][1][r + 8 * hi][ln] = (__bf16)e1[r];
    }
    if (ln == 0) {
#pragma unroll
      for (int r = 0; r < 8; ++r) Xlds[wid][hi * 8 + r] = al[r];
    }
    asm volatile("s_wait_dscnt 0" ::: "memory");
    v16bf Pb = *(const v16bf*)&Plds[wid][hi][ln][0];   // B-fragment of P^T
    float alphaJ = Xlds[wid][ln];                      // alpha, column-indexed
    if (__ballot(alphaJ != 1.0f)) {                    // skip rescale if max unchanged
#pragma unroll
      for (int t = 0; t < 16; ++t) {
#pragma unroll
        for (int r = 0; r < 8; ++r) O[t][r] *= alphaJ;
      }
    }
#pragma unroll
    for (int t = 0; t < 16; ++t) {
      const __bf16* vrow = vb + ((size_t)(t * 16 + ln)) * HW + l0;
      v16bf av = *(const v16bf*)(vrow + kb);
      O[t] = __builtin_amdgcn_wmma_f32_16x16x32_bf16(false, av, false, Pb,
                                                     (short)0, O[t], false, false);
    }
  }

  // final 1/l, column-indexed
  if (ln == 0) {
#pragma unroll
    for (int r = 0; r < 8; ++r) Xlds[wid][hi * 8 + r] = l8[r];
  }
  asm volatile("s_wait_dscnt 0" ::: "memory");
  float linv = 1.0f / Xlds[wid][ln];
  const int j = j0 + ln;
#pragma unroll
  for (int t = 0; t < 16; ++t) {
#pragma unroll
    for (int r = 0; r < 8; r += 2) {
      int c = t * 16 + r + 8 * hi;                     // even
      union { __bf16 h[2]; unsigned u; } pk;
      pk.h[0] = (__bf16)(O[t][r] * linv);
      pk.h[1] = (__bf16)(O[t][r + 1] * linv);
      int blk = 8 + (c >> 5);
      *(unsigned*)(xcat + (((size_t)b * 16 + blk) * HW + j) * 32 + (c & 31)) = pk.u;
    }
  }
}

// ---------------------------------------------------------------------------
extern "C" void kernel_launch(void* const* d_in, const int* in_sizes, int n_in,
                              void* d_out, int out_size, void* d_ws, size_t ws_size,
                              hipStream_t stream) {
  const float* x    = (const float*)d_in[0];
  const float* mask = (const float*)d_in[1];
  const float* Wq   = (const float*)d_in[2];
  const float* bq   = (const float*)d_in[3];
  const float* Wk   = (const float*)d_in[4];
  const float* bk   = (const float*)d_in[5];
  const float* Wv   = (const float*)d_in[6];
  const float* bv   = (const float*)d_in[7];
  const float* Wd   = (const float*)d_in[8];
  const float* Wo   = (const float*)d_in[9];

  char* ws = (char*)d_ws;
  size_t off = 0;
  __bf16* xcat = (__bf16*)(ws + off); off += (size_t)NB * 16 * HW * 32 * 2;   // 33.5 MB
  __bf16* qp   = (__bf16*)(ws + off); off += (size_t)NB * HW * 32 * 2;        // 2.1 MB
  __bf16* kp   = (__bf16*)(ws + off); off += (size_t)NB * HW * 32 * 2;        // 2.1 MB
  __bf16* vcb  = (__bf16*)(ws + off); off += (size_t)NB * 256 * HW * 2;       // 16.8 MB
  float*  dmv  = (float*)(ws + off);  off += (size_t)NB * HW * 4;             // 131 KB
  __bf16* wqp  = (__bf16*)(ws + off); off += (size_t)9 * 8 * 2 * 512 * 2;     // 147 KB
  __bf16* wkp  = (__bf16*)(ws + off); off += (size_t)9 * 8 * 2 * 512 * 2;
  __bf16* wvp  = (__bf16*)(ws + off); off += (size_t)9 * 8 * 16 * 512 * 2;    // 1.18 MB
  __bf16* wop  = (__bf16*)(ws + off); off += (size_t)9 * 16 * 16 * 512 * 2;   // 2.36 MB
  __bf16* zg   = (__bf16*)(ws + off); off += 256;                             // guard

  // 0) zero guard page
  zero_guard_kernel<<<1, 64, 0, stream>>>((unsigned*)zg);
  // 1) cast x into xcat blocks 0..7
  cast_x_kernel<<<32768, 256, 0, stream>>>(x, xcat);
  // 2) repack weights into WMMA fragment layout
  repack_w_kernel<<<18, 256, 0, stream>>>(Wq, wqp, 256, 32);
  repack_w_kernel<<<18, 256, 0, stream>>>(Wk, wkp, 256, 32);
  repack_w_kernel<<<144, 256, 0, stream>>>(Wv, wvp, 256, 256);
  repack_w_kernel<<<288, 256, 0, stream>>>(Wo, wop, 512, 256);
  // 3) visibility mask conv + clamp
  conv_dm_kernel<<<128, 256, 0, stream>>>(mask, Wd, dmv);
  // 4) q, k (channels-last bf16), v (channel-major bf16) convs via WMMA
  //    waves = NB * (Cout/32) * 128 ; 4 waves per block
  conv3x3_wmma<<<256,  128, 0, stream>>>(xcat, wqp, bq, zg, (void*)qp,  8, 2,  0);
  conv3x3_wmma<<<256,  128, 0, stream>>>(xcat, wkp, bk, zg, (void*)kp,  8, 2,  0);
  conv3x3_wmma<<<2048, 128, 0, stream>>>(xcat, wvp, bv, zg, (void*)vcb, 8, 16, 1);
  // 5) fused masked attention -> xcat blocks 8..15
  attn_wmma<<<512, 128, 0, stream>>>(qp, kp, vcb, dmv, xcat);
  // 6) final conv over the 512-channel concat -> fp32 output
  conv3x3_wmma<<<2048, 128, 0, stream>>>(xcat, wop, nullptr, zg, d_out, 16, 16, 2);
}
